// MyInception_11123965297248
// MI455X (gfx1250) — compile-verified, matrix operand order checked by reading, not run on hardware
//
#include <hip/hip_runtime.h>

// ---------------------------------------------------------------------------
// Problem constants (match reference setup_inputs)
// ---------------------------------------------------------------------------
#define PN  200000   // points
#define PC  256      // channels
#define PC4 64       // C/4
#define PC2 128      // C/2
#define PK  27       // kernel offsets
#define PM  60000    // pairs per offset
#define BN_EPS 1e-5f

// ---------------------------------------------------------------------------
// CDNA5 WMMA types
// ---------------------------------------------------------------------------
typedef __attribute__((ext_vector_type(16))) __bf16 v16bf;
typedef __attribute__((ext_vector_type(8)))  float  v8f;

union FragBF {
    v16bf v;
    uint4 u[2];
};

__device__ __forceinline__ unsigned short f32_to_bf16(float f) {
    unsigned int u = __float_as_uint(f);
    // round to nearest even
    u += 0x7FFFu + ((u >> 16) & 1u);
    return (unsigned short)(u >> 16);
}

__device__ __forceinline__ void atomAddF32(float* p, float v) {
    unsafeAtomicAdd(p, v);   // lowers to global_atomic_add_f32 on gfx1250
}

// ---------------------------------------------------------------------------
// Elementwise helpers
// ---------------------------------------------------------------------------
__global__ void k_to_bf16(const float* __restrict__ s, unsigned short* __restrict__ d, int n) {
    int i = blockIdx.x * blockDim.x + threadIdx.x;
    if (i < n) d[i] = f32_to_bf16(s[i]);
}

__global__ void k_zero_f32(float* __restrict__ p, int n) {
    int i = blockIdx.x * blockDim.x + threadIdx.x;
    if (i < n) p[i] = 0.0f;
}

// acc[row][c] = bias[c]  (pre-load bias before scatter-add)
__global__ void k_init_bias(float* __restrict__ p, const float* __restrict__ bias, int n, int C) {
    int i = blockIdx.x * blockDim.x + threadIdx.x;
    if (i < n) p[i] = bias[i % C];
}

// W[K][Cin][Cout] f32  ->  Wt[K][Cout][Cin] bf16
__global__ void k_weight_tr(const float* __restrict__ W, unsigned short* __restrict__ Wt,
                            int K, int Cin, int Cout) {
    int i = blockIdx.x * blockDim.x + threadIdx.x;
    int per = Cin * Cout;
    int total = K * per;
    if (i >= total) return;
    int k = i / per;
    int rem = i - k * per;
    int ci = rem / Cout;
    int co = rem - ci * Cout;
    Wt[(size_t)k * per + (size_t)co * Cin + ci] = f32_to_bf16(W[i]);
}

// ---------------------------------------------------------------------------
// Dense GEMM: C[N x Cout] = A_bf16[N x Cin] @ Wt_bf16[Cout x Cin]^T + bias
// One wave per 16x16 output tile; WMMA f32 <- bf16, K-step 32.
// ---------------------------------------------------------------------------
__global__ void k_gemm_bf16_wmma(const unsigned short* __restrict__ A,
                                 const unsigned short* __restrict__ Bt,
                                 const float* __restrict__ bias,
                                 float* __restrict__ C,
                                 int N, int Cin, int Cout) {
    int gw   = (blockIdx.x * blockDim.x + threadIdx.x) >> 5;
    int lane = threadIdx.x & 31;
    int r    = lane & 15;
    bool hi  = lane >= 16;

    int tilesPerRow = Cout >> 4;
    int tiles = (N >> 4) * tilesPerRow;
    if (gw >= tiles) return;
    int rowTile = gw / tilesPerRow;
    int colTile = gw - rowTile * tilesPerRow;

    const unsigned short* arow = A  + (size_t)(rowTile * 16 + r) * Cin;
    const unsigned short* brow = Bt + (size_t)(colTile * 16 + r) * Cin;

    v8f acc = {};
    int kofs = hi ? 8 : 0;
    for (int kk = 0; kk < Cin; kk += 32) {
        FragBF a, b;
        int k0 = kk + kofs;
        a.u[0] = *(const uint4*)(arow + k0);
        a.u[1] = *(const uint4*)(arow + k0 + 16);
        b.u[0] = *(const uint4*)(brow + k0);
        b.u[1] = *(const uint4*)(brow + k0 + 16);
        acc = __builtin_amdgcn_wmma_f32_16x16x32_bf16(false, a.v, false, b.v,
                                                      (short)0, acc, false, false);
    }

    int col = colTile * 16 + r;           // N index == lane%16
    float bv = bias[col];
    int mbase = rowTile * 16 + (hi ? 8 : 0);
#pragma unroll
    for (int i = 0; i < 8; ++i)
        C[(size_t)(mbase + i) * Cout + col] = acc[i] + bv;
}

// ---------------------------------------------------------------------------
// Sparse conv (gather -> WMMA -> atomic scatter-add)
// grid.x = K * (M/16); block = 32 * (Cout/16) threads, one wave per col tile.
// ---------------------------------------------------------------------------
__global__ void k_sparse_conv_wmma(const unsigned short* __restrict__ src,   // [N x Cin] bf16
                                   const unsigned short* __restrict__ Wt,    // [K x Cout x Cin] bf16
                                   const int* __restrict__ in_idx,           // [K x M]
                                   const int* __restrict__ out_idx,          // [K x M]
                                   float* __restrict__ acc,                  // [N x Cout] f32
                                   int Cin, int Cout, int M, int mtiles) {
    __shared__ int s_in[16];
    __shared__ int s_out[16];

    int k  = blockIdx.x / mtiles;
    int mt = blockIdx.x - k * mtiles;
    int t  = threadIdx.x;
    if (t < 16)       s_in[t]       = in_idx [(size_t)k * M + mt * 16 + t];
    else if (t < 32)  s_out[t - 16] = out_idx[(size_t)k * M + mt * 16 + (t - 16)];
    __syncthreads();

    int wave = t >> 5;
    int lane = t & 31;
    int r    = lane & 15;
    bool hi  = lane >= 16;

    const unsigned short* arow = src + (size_t)s_in[r] * Cin;
    const unsigned short* brow = Wt + ((size_t)k * Cout + wave * 16 + r) * Cin;

    v8f c = {};
    int kofs = hi ? 8 : 0;
    for (int kk = 0; kk < Cin; kk += 32) {
        FragBF a, b;
        int k0 = kk + kofs;
        a.u[0] = *(const uint4*)(arow + k0);
        a.u[1] = *(const uint4*)(arow + k0 + 16);
        b.u[0] = *(const uint4*)(brow + k0);
        b.u[1] = *(const uint4*)(brow + k0 + 16);
        c = __builtin_amdgcn_wmma_f32_16x16x32_bf16(false, a.v, false, b.v,
                                                    (short)0, c, false, false);
    }

    int co = wave * 16 + r;
    int mofs = hi ? 8 : 0;
#pragma unroll
    for (int i = 0; i < 8; ++i) {
        int orow = s_out[i + mofs];
        atomAddF32(acc + (size_t)orow * Cout + co, c[i]);
    }
}

// ---------------------------------------------------------------------------
// BatchNorm statistics: per-channel sum and sum-of-squares over N rows.
// stats[0..C) = sum, stats[C..2C) = sumsq. blockDim must be multiple of C.
// ---------------------------------------------------------------------------
__global__ void k_bn_stats(const float* __restrict__ x, float* __restrict__ stats,
                           int N, int C) {
    __shared__ float sh1[256];
    __shared__ float sh2[256];
    int t = threadIdx.x;
    int c = t % C;
    int rep = t / C;
    int nrep = blockDim.x / C;

    float s = 0.0f, s2 = 0.0f;
    for (int row = blockIdx.x * nrep + rep; row < N; row += gridDim.x * nrep) {
        float v = x[(size_t)row * C + c];
        s  += v;
        s2 += v * v;
    }
    sh1[t] = s;
    sh2[t] = s2;
    __syncthreads();
    if (rep == 0) {
        for (int j = 1; j < nrep; ++j) {
            s  += sh1[c + j * C];
            s2 += sh2[c + j * C];
        }
        atomAddF32(stats + c, s);
        atomAddF32(stats + C + c, s2);
    }
}

// ---------------------------------------------------------------------------
// BN(training stats) + ReLU, emit bf16 for the next gather stage.
// ---------------------------------------------------------------------------
__global__ void k_bnrelu_bf16(const float* __restrict__ src, const float* __restrict__ stats,
                              const float* __restrict__ gamma, const float* __restrict__ beta,
                              unsigned short* __restrict__ dst, int n, int C, float invN) {
    int i = blockIdx.x * blockDim.x + threadIdx.x;
    if (i >= n) return;
    int ch = i % C;
    float mu  = stats[ch] * invN;
    float var = stats[C + ch] * invN - mu * mu;
    float y = (src[i] - mu) * rsqrtf(var + BN_EPS) * gamma[ch] + beta[ch];
    dst[i] = f32_to_bf16(fmaxf(y, 0.0f));
}

// BN + ReLU + residual add of x, written into the concat slot of d_out (f32).
__global__ void k_bnrelu_residual(const float* __restrict__ src, const float* __restrict__ stats,
                                  const float* __restrict__ gamma, const float* __restrict__ beta,
                                  const float* __restrict__ x, float* __restrict__ out,
                                  int N, int C, int colOff, int Cfull, float invN) {
    int i = blockIdx.x * blockDim.x + threadIdx.x;
    if (i >= N * C) return;
    int row = i / C;
    int ch  = i - row * C;
    float mu  = stats[ch] * invN;
    float var = stats[C + ch] * invN - mu * mu;
    float y = (src[i] - mu) * rsqrtf(var + BN_EPS) * gamma[ch] + beta[ch];
    size_t oi = (size_t)row * Cfull + colOff + ch;
    out[oi] = fmaxf(y, 0.0f) + x[oi];
}

// ---------------------------------------------------------------------------
// Host orchestration
// ---------------------------------------------------------------------------
static inline int cdiv(long long a, long long b) { return (int)((a + b - 1) / b); }

extern "C" void kernel_launch(void* const* d_in, const int* in_sizes, int n_in,
                              void* d_out, int out_size, void* d_ws, size_t ws_size,
                              hipStream_t stream) {
    (void)in_sizes; (void)n_in; (void)out_size; (void)ws_size;

    const float* x       = (const float*)d_in[0];
    const int*   in_idx  = (const int*)  d_in[1];
    const int*   out_idx = (const int*)  d_in[2];
    const float* W1 = (const float*)d_in[3];  const float* b1 = (const float*)d_in[4];
    const float* g1 = (const float*)d_in[5];  const float* be1= (const float*)d_in[6];
    const float* W2 = (const float*)d_in[7];  const float* b2 = (const float*)d_in[8];
    const float* g2 = (const float*)d_in[9];  const float* be2= (const float*)d_in[10];
    const float* W3 = (const float*)d_in[11]; const float* b3 = (const float*)d_in[12];
    const float* g3 = (const float*)d_in[13]; const float* be3= (const float*)d_in[14];
    const float* W4 = (const float*)d_in[15]; const float* b4 = (const float*)d_in[16];
    const float* g4 = (const float*)d_in[17]; const float* be4= (const float*)d_in[18];
    const float* W5 = (const float*)d_in[19]; const float* b5 = (const float*)d_in[20];
    const float* g5 = (const float*)d_in[21]; const float* be5= (const float*)d_in[22];
    float* out = (float*)d_out;

    // ---- workspace carve-up (aligned 256B) ----
    char* base = (char*)d_ws;
    size_t off = 0;
    auto carve = [&](size_t bytes) -> char* {
        char* p = base + off;
        off = (off + bytes + 255) & ~(size_t)255;
        return p;
    };
    unsigned short* x_bf  = (unsigned short*)carve((size_t)PN * PC  * 2);  // x in bf16
    float*          t1    = (float*)         carve((size_t)PN * PC4 * 4);  // conv1 out / u1 acc
    unsigned short* t1b   = (unsigned short*)carve((size_t)PN * PC4 * 2);  // bn(t1) bf16 (shared)
    float*          t2    = (float*)         carve((size_t)PN * PC4 * 4);  // conv2 acc
    unsigned short* t2b   = (unsigned short*)carve((size_t)PN * PC4 * 2);  // bn(t2) bf16
    float*          t3    = (float*)         carve((size_t)PN * PC2 * 4);  // conv3 out / u2 acc
    unsigned short* W1t   = (unsigned short*)carve((size_t)PC  * PC4 * 2);
    unsigned short* W2t   = (unsigned short*)carve((size_t)PK * PC4 * PC4 * 2);
    unsigned short* W3t   = (unsigned short*)carve((size_t)PC4 * PC2 * 2);
    unsigned short* W4t   = (unsigned short*)carve((size_t)PK * PC  * PC4 * 2);
    unsigned short* W5t   = (unsigned short*)carve((size_t)PK * PC4 * PC2 * 2);
    float*          stats = (float*)         carve(2 * 128 * 4);

    const float invN = 1.0f / (float)PN;
    const int TB = 256;
    const int mtiles = PM / 16;                 // 3750
    const int sparseGrid = PK * mtiles;         // 101250

    // ---- precision conversion / weight transpose ----
    k_to_bf16   <<<cdiv((long long)PN * PC, TB), TB, 0, stream>>>(x, x_bf, PN * PC);
    k_weight_tr <<<cdiv((long long)PC * PC4, TB), TB, 0, stream>>>(W1, W1t, 1,  PC,  PC4);
    k_weight_tr <<<cdiv((long long)PK * PC4 * PC4, TB), TB, 0, stream>>>(W2, W2t, PK, PC4, PC4);
    k_weight_tr <<<cdiv((long long)PC4 * PC2, TB), TB, 0, stream>>>(W3, W3t, 1,  PC4, PC2);
    k_weight_tr <<<cdiv((long long)PK * PC  * PC4, TB), TB, 0, stream>>>(W4, W4t, PK, PC,  PC4);
    k_weight_tr <<<cdiv((long long)PK * PC4 * PC2, TB), TB, 0, stream>>>(W5, W5t, PK, PC4, PC2);

    // ================= Branch 1 =================
    // conv1: t1 = x @ W1 + b1
    {
        int tiles = (PN / 16) * (PC4 / 16);
        k_gemm_bf16_wmma<<<cdiv(tiles, 8), 256, 0, stream>>>(x_bf, W1t, b1, t1, PN, PC, PC4);
    }
    k_zero_f32 <<<1, 256, 0, stream>>>(stats, 2 * PC4);
    k_bn_stats <<<1024, 256, 0, stream>>>(t1, stats, PN, PC4);
    k_bnrelu_bf16<<<cdiv((long long)PN * PC4, TB), TB, 0, stream>>>(t1, stats, g1, be1, t1b, PN * PC4, PC4, invN);

    // conv2 (sparse): t2 = scatter(gather(t1b) @ W2) + b2
    k_init_bias<<<cdiv((long long)PN * PC4, TB), TB, 0, stream>>>(t2, b2, PN * PC4, PC4);
    k_sparse_conv_wmma<<<sparseGrid, 32 * (PC4 / 16), 0, stream>>>(t1b, W2t, in_idx, out_idx, t2, PC4, PC4, PM, mtiles);
    k_zero_f32 <<<1, 256, 0, stream>>>(stats, 2 * PC4);
    k_bn_stats <<<1024, 256, 0, stream>>>(t2, stats, PN, PC4);
    k_bnrelu_bf16<<<cdiv((long long)PN * PC4, TB), TB, 0, stream>>>(t2, stats, g2, be2, t2b, PN * PC4, PC4, invN);

    // conv3: t3 = t2b @ W3 + b3
    {
        int tiles = (PN / 16) * (PC2 / 16);
        k_gemm_bf16_wmma<<<cdiv(tiles, 8), 256, 0, stream>>>(t2b, W3t, b3, t3, PN, PC4, PC2);
    }
    k_zero_f32 <<<1, 256, 0, stream>>>(stats, 2 * PC2);
    k_bn_stats <<<1024, 256, 0, stream>>>(t3, stats, PN, PC2);
    k_bnrelu_residual<<<cdiv((long long)PN * PC2, TB), TB, 0, stream>>>(t3, stats, g3, be3, x, out, PN, PC2, 0, PC, invN);

    // ================= Branch 2 =================
    // conv4 (sparse, Cin=256): u1 (reuse t1) = scatter(gather(x_bf) @ W4) + b4
    k_init_bias<<<cdiv((long long)PN * PC4, TB), TB, 0, stream>>>(t1, b4, PN * PC4, PC4);
    k_sparse_conv_wmma<<<sparseGrid, 32 * (PC4 / 16), 0, stream>>>(x_bf, W4t, in_idx, out_idx, t1, PC, PC4, PM, mtiles);
    k_zero_f32 <<<1, 256, 0, stream>>>(stats, 2 * PC4);
    k_bn_stats <<<1024, 256, 0, stream>>>(t1, stats, PN, PC4);
    k_bnrelu_bf16<<<cdiv((long long)PN * PC4, TB), TB, 0, stream>>>(t1, stats, g4, be4, t1b, PN * PC4, PC4, invN);

    // conv5 (sparse, Cout=128): u2 (reuse t3) = scatter(gather(t1b) @ W5) + b5
    k_init_bias<<<cdiv((long long)PN * PC2, TB), TB, 0, stream>>>(t3, b5, PN * PC2, PC2);
    k_sparse_conv_wmma<<<sparseGrid, 32 * (PC2 / 16), 0, stream>>>(t1b, W5t, in_idx, out_idx, t3, PC4, PC2, PM, mtiles);
    k_zero_f32 <<<1, 256, 0, stream>>>(stats, 2 * PC2);
    k_bn_stats <<<1024, 256, 0, stream>>>(t3, stats, PN, PC2);
    k_bnrelu_residual<<<cdiv((long long)PN * PC2, TB), TB, 0, stream>>>(t3, stats, g5, be5, x, out, PN, PC2, PC2, PC, invN);
}
